// MultiHeadAttention_84043920048297
// MI455X (gfx1250) — compile-verified, matrix operand order checked by reading, not run on hardware
//
#include <hip/hip_runtime.h>
#include <hip/hip_bf16.h>

// ---------------------------------------------------------------------------
// MI455X (gfx1250) ViT relative-position attention, bf16 WMMA throughout.
//   B=16, H=12, N=577 (pad 608), d=64, C=768.
// Pipeline:
//   1) convert x / qkv_w / proj_w to bf16
//   2) QKV GEMM (wmma bf16, software-pipelined fragment loads)
//   3) r_q[n, 0..99] = q[n,:32]@q_x_emb.T , q[n,32:]@q_y_emb.T  (tiny)
//   4) fused attention, 2 waves per (bh, 16-row tile):
//        S = Q K^T (wmma, m-tiles split across waves, double-buffered K),
//        bias via r_q gathers, exact 2-pass softmax in LDS (4 lanes/row),
//        per-row 50-bin rel-offset histograms (ds_add_f32),
//        O = P V (wmma, m-chunks split across waves, combined via LDS),
//        + hist @ v_{x,y}_emb, normalize, store bf16
//   5) projection GEMM (wmma bf16) -> fp32 d_out
// ---------------------------------------------------------------------------

typedef __bf16 bf16_t;
typedef __attribute__((ext_vector_type(16))) __bf16 v16bf;
typedef __attribute__((ext_vector_type(8)))  __bf16 v8bf;
typedef __attribute__((ext_vector_type(8)))  float  v8f;

#define B_    16
#define NH_   12
#define N_    577
#define NPAD_ 608      // 38 * 16
#define D_    64
#define C_    768
#define QKV_  2304
#define SCALE_ 0.125f  // 64^-0.5
#define SW_   616      // padded LDS row stride for S (bank-conflict pad)

static __device__ __forceinline__ v8f wmma_bf16(v16bf a, v16bf b, v8f c) {
  return __builtin_amdgcn_wmma_f32_16x16x32_bf16(false, a, false, b, (short)0, c,
                                                 false, false);
}

// 16-bit WMMA fragment (16x32) from a row-major matrix where the K dimension
// is contiguous within each row.  Lane L: row = row0 + L%16, K-halves start at
// 8*(L/16) and 8*(L/16)+16 -> two contiguous 16-byte loads.
static __device__ __forceinline__ v16bf load_frag(const bf16_t* base, int stride,
                                                  int row0, int lane) {
  const int lo = lane & 15, hi = (lane >> 4) & 1;
  const bf16_t* p = base + (size_t)(row0 + lo) * (size_t)stride + hi * 8;
  v8bf a = *(const v8bf*)p;
  v8bf b = *(const v8bf*)(p + 16);
  v16bf r;
#pragma unroll
  for (int i = 0; i < 8; ++i) { r[i] = a[i]; r[8 + i] = b[i]; }
  return r;
}

// ---------------------------------------------------------------------------
__global__ __launch_bounds__(256) void cvt_bf16_kernel(const float* __restrict__ in,
                                                       bf16_t* __restrict__ out,
                                                       size_t n) {
  size_t i = (size_t)blockIdx.x * blockDim.x + threadIdx.x;
  const size_t stride = (size_t)gridDim.x * blockDim.x;
  for (; i < n; i += stride) out[i] = (bf16_t)in[i];
}

__global__ __launch_bounds__(256) void zero32_kernel(uint32_t* __restrict__ p,
                                                     size_t nwords) {
  size_t i = (size_t)blockIdx.x * blockDim.x + threadIdx.x;
  const size_t stride = (size_t)gridDim.x * blockDim.x;
  for (; i < nwords; i += stride) p[i] = 0u;
}

// ---------------------------------------------------------------------------
// Software-pipelined 16x64-per-wave bf16 WMMA GEMM body.
// A: (rows x 768) row-major, B rows are output channels (w layout).
template <bool kQkvScatter>
static __device__ __forceinline__ void gemm_body(
    const bf16_t* __restrict__ amat, const bf16_t* __restrict__ wmat,
    int rowTile, int colBase, int lane,
    bf16_t* __restrict__ q, bf16_t* __restrict__ k, bf16_t* __restrict__ v,
    float* __restrict__ fout) {
  const int lo = lane & 15, hi = (lane >> 4) & 1;
  v8f acc0 = {}, acc1 = {}, acc2 = {}, acc3 = {};

  // prologue: fragments for kc = 0
  v16bf a  = load_frag(amat + 0, C_, rowTile * 16, lane);
  v16bf b0 = load_frag(wmat + 0, C_, colBase +  0, lane);
  v16bf b1 = load_frag(wmat + 0, C_, colBase + 16, lane);
  v16bf b2 = load_frag(wmat + 0, C_, colBase + 32, lane);
  v16bf b3 = load_frag(wmat + 0, C_, colBase + 48, lane);

  for (int kc = 0; kc < C_; kc += 32) {
    const int kn = kc + 32;
    v16bf an = a, n0 = b0, n1 = b1, n2 = b2, n3 = b3;
    if (kn < C_) {  // issue next-iteration loads before consuming current
      an = load_frag(amat + kn, C_, rowTile * 16, lane);
      n0 = load_frag(wmat + kn, C_, colBase +  0, lane);
      n1 = load_frag(wmat + kn, C_, colBase + 16, lane);
      n2 = load_frag(wmat + kn, C_, colBase + 32, lane);
      n3 = load_frag(wmat + kn, C_, colBase + 48, lane);
    }
    if (kn + 64 < C_) {  // warm L2 two chunks ahead (global_prefetch_b8)
      __builtin_prefetch(amat + (size_t)(rowTile * 16 + lo) * C_ + kn + 64, 0, 0);
      __builtin_prefetch(wmat + (size_t)(colBase + lo) * C_ + kn + 64, 0, 0);
    }
    acc0 = wmma_bf16(a, b0, acc0);
    acc1 = wmma_bf16(a, b1, acc1);
    acc2 = wmma_bf16(a, b2, acc2);
    acc3 = wmma_bf16(a, b3, acc3);
    a = an; b0 = n0; b1 = n1; b2 = n2; b3 = n3;
  }

  v8f accs[4] = {acc0, acc1, acc2, acc3};
  if (kQkvScatter) {
    bf16_t* bufs[3] = {q, k, v};
#pragma unroll
    for (int c = 0; c < 4; ++c) {
      const int col = colBase + c * 16 + lo;
      const int which = col / C_;
      const int hh = (col % C_) / D_;
      const int dd = col % D_;
      bf16_t* dst = bufs[which];
#pragma unroll
      for (int r = 0; r < 8; ++r) {
        const int row = rowTile * 16 + r + 8 * hi;   // 0..9231
        const int bb = row / N_;
        const int nn = row % N_;
        dst[(((size_t)bb * NH_ + hh) * NPAD_ + nn) * D_ + dd] = (bf16_t)accs[c][r];
      }
    }
  } else {
#pragma unroll
    for (int c = 0; c < 4; ++c) {
      const int col = colBase + c * 16 + lo;
#pragma unroll
      for (int r = 0; r < 8; ++r) {
        const int row = rowTile * 16 + r + 8 * hi;
        fout[(size_t)row * C_ + col] = accs[c][r];
      }
    }
  }
}

// QKV GEMM: (9232 x 768) @ (768 x 2304), scatter into q/k/v (B*H, NPAD, 64).
__global__ __launch_bounds__(256) void gemm_qkv_kernel(
    const bf16_t* __restrict__ xb, const bf16_t* __restrict__ wb,
    bf16_t* __restrict__ q, bf16_t* __restrict__ k, bf16_t* __restrict__ v) {
  const int lane = threadIdx.x & 31;
  const int wave = threadIdx.x >> 5;
  const int rowTile = blockIdx.x * 8 + wave;   // 577 tiles of 16 rows
  if (rowTile >= N_) return;                   // wave-uniform exit
  gemm_body<true>(xb, wb, rowTile, blockIdx.y * 64, lane, q, k, v, nullptr);
}

// Projection GEMM: (9232 x 768) @ (768 x 768) -> fp32 out.
__global__ __launch_bounds__(256) void gemm_proj_kernel(
    const bf16_t* __restrict__ ab, const bf16_t* __restrict__ wb,
    float* __restrict__ out) {
  const int lane = threadIdx.x & 31;
  const int wave = threadIdx.x >> 5;
  const int rowTile = blockIdx.x * 8 + wave;
  if (rowTile >= N_) return;
  gemm_body<false>(ab, wb, rowTile, blockIdx.y * 64, lane, nullptr, nullptr,
                   nullptr, out);
}

// ---------------------------------------------------------------------------
// r_q[bh][n][i] : i<50 -> q[:32]@q_x_emb[i], i in 50..99 -> q[32:]@q_y_emb[i-50]
__global__ __launch_bounds__(128) void relq_kernel(
    const bf16_t* __restrict__ qb, const float* __restrict__ qx,
    const float* __restrict__ qy, float* __restrict__ rq) {
  __shared__ float qrow[64];
  const int n = blockIdx.x;       // 0..NPAD_-1 (padded rows are zero -> rq 0)
  const int bh = blockIdx.y;
  const int tid = threadIdx.x;
  if (tid < 64) qrow[tid] = (float)qb[((size_t)bh * NPAD_ + n) * D_ + tid];
  __syncthreads();
  if (tid < 100) {
    float s = 0.f;
    if (tid < 50) {
      const float* e = qx + tid * 32;
#pragma unroll
      for (int t = 0; t < 32; ++t) s += qrow[t] * e[t];
    } else {
      const float* e = qy + (tid - 50) * 32;
#pragma unroll
      for (int t = 0; t < 32; ++t) s += qrow[32 + t] * e[t];
    }
    rq[((size_t)bh * NPAD_ + n) * 100 + tid] = s;
  }
}

// ---------------------------------------------------------------------------
// Fused attention: 2 waves per (bh, 16-row tile).
__global__ __launch_bounds__(64) void attn_kernel(
    const bf16_t* __restrict__ qb, const bf16_t* __restrict__ kb,
    const bf16_t* __restrict__ vb, const float* __restrict__ rq,
    const float* __restrict__ vx, const float* __restrict__ vy,
    bf16_t* __restrict__ aout) {
  __shared__ float  sbuf[16 * SW_];     // full S rows (then P rows)
  __shared__ bf16_t vt[2][64][32];      // per-wave transposed V tile [d][m]
  __shared__ float  hist[16][104];      // rel-offset histograms (x:0..49, y:50..99)
  __shared__ float  linv[16];
  __shared__ float  obuf[16][64];       // combined O tile

  const int tid  = threadIdx.x;
  const int wave = tid >> 5;
  const int lane = tid & 31;
  const int lo = lane & 15, hi = (lane >> 4) & 1;
  const int bh = blockIdx.y;
  const int b = bh / NH_, h = bh % NH_;
  const int nBase = blockIdx.x * 16;

  const bf16_t* qrow = qb + (size_t)bh * NPAD_ * D_;
  const bf16_t* krow = kb + (size_t)bh * NPAD_ * D_;
  const bf16_t* vrow = vb + (size_t)bh * NPAD_ * D_;

  const v16bf aq0 = load_frag(qrow +  0, D_, nBase, lane);
  const v16bf aq1 = load_frag(qrow + 32, D_, nBase, lane);

  // ---- Phase 1: S = Q K^T, m-tiles interleaved across the 2 waves,
  //      double-buffered K fragments -----------------------------------
  {
    int mt = wave;
    v16bf bk0 = load_frag(krow +  0, D_, mt * 16, lane);
    v16bf bk1 = load_frag(krow + 32, D_, mt * 16, lane);
    while (mt < NPAD_ / 16) {
      const int mtn = mt + 2;
      v16bf nb0 = bk0, nb1 = bk1;
      if (mtn < NPAD_ / 16) {  // prefetch next tile before the WMMAs
        nb0 = load_frag(krow +  0, D_, mtn * 16, lane);
        nb1 = load_frag(krow + 32, D_, mtn * 16, lane);
      }
      v8f acc = {};
      acc = wmma_bf16(aq0, bk0, acc);
      acc = wmma_bf16(aq1, bk1, acc);
#pragma unroll
      for (int r = 0; r < 8; ++r)
        sbuf[(r + 8 * hi) * SW_ + mt * 16 + lo] = acc[r];
      bk0 = nb0; bk1 = nb1; mt = mtn;
    }
  }
  for (int i = tid; i < 16 * 104; i += 64) (&hist[0][0])[i] = 0.f;
  __syncthreads();

  // ---- Phase 2: bias + exact softmax + histograms (4 lanes per row) ----
  {
    const int row = tid >> 2;
    const int par = tid & 3;
    const int nglob = nBase + row;
    const bool rowOK = (nglob < N_);
    const float* rqn = rq + ((size_t)bh * NPAD_ + (rowOK ? nglob : 0)) * 100;
    int xn = 0, yn = 0;
    if (nglob > 0) { int p = nglob - 1; xn = p % 24; yn = p / 24; }

    float mx = -1e30f;
    for (int m = par; m < NPAD_; m += 4) {
      float s = -1e30f;
      if (rowOK && m < N_) {
        s = sbuf[row * SW_ + m];
        int ix = 0, iy = 0;
        if (nglob > 0 && m > 0) {
          const int pm = m - 1;
          int dx = (pm % 24) - xn; dx = dx < -24 ? -24 : (dx > 24 ? 24 : dx);
          int dy = (pm / 24) - yn; dy = dy < -24 ? -24 : (dy > 24 ? 24 : dy);
          ix = dx + 25; iy = dy + 25;
        }
        s = (s + rqn[ix] + rqn[50 + iy]) * SCALE_;
      }
      sbuf[row * SW_ + m] = s;
      mx = fmaxf(mx, s);
    }
    mx = fmaxf(mx, __shfl_xor(mx, 1, 32));
    mx = fmaxf(mx, __shfl_xor(mx, 2, 32));

    float sum = 0.f;
    for (int m = par; m < NPAD_; m += 4) {
      const float s = sbuf[row * SW_ + m];
      float p = 0.f;
      if (s > -1e29f) {
        p = __expf(s - mx);
        int ix = 0, iy = 0;
        if (nglob > 0 && m > 0) {
          const int pm = m - 1;
          int dx = (pm % 24) - xn; dx = dx < -24 ? -24 : (dx > 24 ? 24 : dx);
          int dy = (pm / 24) - yn; dy = dy < -24 ? -24 : (dy > 24 ? 24 : dy);
          ix = dx + 25; iy = dy + 25;
        }
        atomicAdd(&hist[row][ix], p);          // ds_add_f32
        atomicAdd(&hist[row][50 + iy], p);
        sum += p;
      }
      sbuf[row * SW_ + m] = p;
    }
    sum += __shfl_xor(sum, 1, 32);
    sum += __shfl_xor(sum, 2, 32);
    if (par == 0) linv[row] = (sum > 0.f) ? (1.f / sum) : 0.f;
  }
  __syncthreads();

  // ---- Phase 3: O = P V, m-chunks interleaved across the 2 waves --------
  v8f o[4] = {};
  for (int mc = wave; mc < NPAD_ / 32; mc += 2) {
    const int mBase = mc * 32;
    {  // stage transposed V tile for this wave: vt[wave][d][m]
      const bf16_t* vr = vrow + (size_t)(mBase + lane) * D_;
#pragma unroll
      for (int dc = 0; dc < 8; ++dc) {
        v8bf chunk = *(const v8bf*)(vr + dc * 8);
#pragma unroll
        for (int i = 0; i < 8; ++i) vt[wave][dc * 8 + i][lane] = chunk[i];
      }
    }
    v16bf pf;
    {
      const float* sp = &sbuf[lo * SW_ + mBase + hi * 8];
#pragma unroll
      for (int i = 0; i < 8; ++i) { pf[i] = (bf16_t)sp[i]; pf[8 + i] = (bf16_t)sp[16 + i]; }
    }
#pragma unroll
    for (int c = 0; c < 4; ++c) {
      const bf16_t* vp = &vt[wave][c * 16 + lo][hi * 8];
      v16bf bfrag;
#pragma unroll
      for (int i = 0; i < 8; ++i) { bfrag[i] = vp[i]; bfrag[8 + i] = vp[16 + i]; }
      o[c] = wmma_bf16(pf, bfrag, o[c]);
    }
  }
  __syncthreads();

  // combine the two partial O tiles through LDS
  if (wave == 0) {
#pragma unroll
    for (int c = 0; c < 4; ++c)
#pragma unroll
      for (int r = 0; r < 8; ++r) obuf[r + 8 * hi][c * 16 + lo] = o[c][r];
  }
  __syncthreads();
  if (wave == 1) {
#pragma unroll
    for (int c = 0; c < 4; ++c)
#pragma unroll
      for (int r = 0; r < 8; ++r) obuf[r + 8 * hi][c * 16 + lo] += o[c][r];
  }
  __syncthreads();

  // ---- Phase 4: + hist @ v_pos embeddings, normalize, store --------------
  // wave 0 handles c=0,1 (x-embedding half), wave 1 handles c=2,3 (y half).
  float li[8];
#pragma unroll
  for (int r = 0; r < 8; ++r) li[r] = linv[r + 8 * hi];
#pragma unroll
  for (int cc = 0; cc < 2; ++cc) {
    const int c = wave * 2 + cc;
    const int dcol = c * 16 + lo;
    const float* emb = (c < 2) ? (vx + dcol) : (vy + (dcol - 32));
    const int hoff = (c < 2) ? 0 : 50;
    float vp[8] = {0.f, 0.f, 0.f, 0.f, 0.f, 0.f, 0.f, 0.f};
    for (int i = 0; i < 50; ++i) {
      const float e = emb[i * 32];
#pragma unroll
      for (int r = 0; r < 8; ++r) vp[r] += hist[r + 8 * hi][hoff + i] * e;
    }
#pragma unroll
    for (int r = 0; r < 8; ++r) {
      const int nglob = nBase + r + 8 * hi;
      if (nglob < N_) {
        const size_t idx = ((size_t)(b * N_ + nglob)) * C_ + (size_t)h * D_ + dcol;
        aout[idx] = (bf16_t)((obuf[r + 8 * hi][dcol] + vp[r]) * li[r]);
      }
    }
  }
}

// ---------------------------------------------------------------------------
extern "C" void kernel_launch(void* const* d_in, const int* in_sizes, int n_in,
                              void* d_out, int out_size, void* d_ws, size_t ws_size,
                              hipStream_t stream) {
  (void)in_sizes; (void)n_in; (void)out_size; (void)ws_size;
  const float* x      = (const float*)d_in[0];
  const float* qkv_w  = (const float*)d_in[1];
  const float* proj_w = (const float*)d_in[2];
  const float* q_x    = (const float*)d_in[3];
  const float* q_y    = (const float*)d_in[4];
  const float* v_x    = (const float*)d_in[5];
  const float* v_y    = (const float*)d_in[6];

  char* ws = (char*)d_ws;
  const size_t ROWS      = (size_t)B_ * N_;                 // 9232
  const size_t SZ_XB     = ROWS * C_ * 2;                   // bf16 x
  const size_t SZ_WQKV   = (size_t)QKV_ * C_ * 2;
  const size_t SZ_WPROJ  = (size_t)C_ * C_ * 2;
  const size_t SZ_QKVBUF = (size_t)B_ * NH_ * NPAD_ * D_ * 2;
  const size_t SZ_RQ     = (size_t)B_ * NH_ * NPAD_ * 100 * 4;

  size_t off = 0;
  bf16_t* xb    = (bf16_t*)(ws + off); off += SZ_XB;
  bf16_t* wqkv  = (bf16_t*)(ws + off); off += SZ_WQKV;
  bf16_t* wproj = (bf16_t*)(ws + off); off += SZ_WPROJ;
  bf16_t* qbuf  = (bf16_t*)(ws + off); off += SZ_QKVBUF;
  bf16_t* kbuf  = (bf16_t*)(ws + off); off += SZ_QKVBUF;
  bf16_t* vbuf  = (bf16_t*)(ws + off); off += SZ_QKVBUF;
  float*  rqbuf = (float*)(ws + off);  off += SZ_RQ;
  bf16_t* abuf  = (bf16_t*)(ws + off); off += SZ_XB;

  cvt_bf16_kernel<<<2048, 256, 0, stream>>>(x, xb, ROWS * C_);
  cvt_bf16_kernel<<<1024, 256, 0, stream>>>(qkv_w, wqkv, (size_t)QKV_ * C_);
  cvt_bf16_kernel<<<512, 256, 0, stream>>>(proj_w, wproj, (size_t)C_ * C_);
  zero32_kernel<<<2048, 256, 0, stream>>>((uint32_t*)qbuf, 3 * SZ_QKVBUF / 4);

  gemm_qkv_kernel<<<dim3(73, 36), 256, 0, stream>>>(xb, wqkv, qbuf, kbuf, vbuf);
  relq_kernel<<<dim3(NPAD_, B_ * NH_), 128, 0, stream>>>(qbuf, q_x, q_y, rqbuf);
  attn_kernel<<<dim3(37, B_ * NH_), 64, 0, stream>>>(qbuf, kbuf, vbuf, rqbuf,
                                                     v_x, v_y, abuf);
  gemm_proj_kernel<<<dim3(73, 12), 256, 0, stream>>>(abuf, wproj, (float*)d_out);
}
